// GRU_9715216024223
// MI455X (gfx1250) — matrix operand order repaired and checked by reading, not compiled, not run on hardware
//
#include <hip/hip_runtime.h>
#include <hip/hip_bf16.h>

// Problem constants (from the reference)
#define T_STEPS 128
#define BATCH   32
#define EDIM    1024
#define HDIM    1024
#define VOCAB   32000
#define NLAYER  2
#define TILE_K  32

typedef __bf16 bf16_t;
typedef __attribute__((ext_vector_type(16))) __bf16 v16bf;
typedef __attribute__((ext_vector_type(8)))  __bf16 v8bf;
typedef __attribute__((ext_vector_type(8)))  float  v8f;

// ---------------------------------------------------------------------------
// Helpers
// ---------------------------------------------------------------------------

__device__ __forceinline__ bf16_t f32_to_bf16_rn(float f) {
  union { float f; unsigned u; } v; v.f = f;
  unsigned u = v.u;
  if ((u & 0x7F800000u) == 0x7F800000u) {           // inf / nan
    unsigned short h = (unsigned short)(u >> 16);
    if (u & 0x007FFFFFu) h |= 0x0040;               // quiet the nan
    return __builtin_bit_cast(bf16_t, h);
  }
  u += 0x7FFFu + ((u >> 16) & 1u);                  // round to nearest even
  unsigned short h = (unsigned short)(u >> 16);
  return __builtin_bit_cast(bf16_t, h);
}

__device__ __forceinline__ float sigmoid_f(float x) {
  return 1.0f / (1.0f + __expf(-x));
}
__device__ __forceinline__ float tanh_f(float x) {
  return 2.0f * sigmoid_f(2.0f * x) - 1.0f;
}

// A-matrix fragment (16x32 bf16, row-major source, ld in elements).
// ISA layout: lane L<16 holds row L, K {0..7} in regs 0-3 and K {16..23} in 4-7;
// lane L>=16 holds row L-16, K {8..15} and {24..31}.
__device__ __forceinline__ v16bf load_a_frag(const bf16_t* __restrict__ tile,
                                             int ld, int lane) {
  const int half = lane >> 4;
  const int r    = lane & 15;
  const bf16_t* p = tile + (size_t)r * ld + half * 8;
  v8bf lo = *(const v8bf*)(p);
  v8bf hi = *(const v8bf*)(p + 16);
  v16bf out;
#pragma unroll
  for (int i = 0; i < 8; ++i) { out[i] = lo[i]; out[i + 8] = hi[i]; }
  return out;
}

// B-matrix fragment for A@W^T where W is row-major N x K (K contiguous).
// WMMA's B is K x 16: lane holds column (lane&15), K {0..7}+(lane>>4)*8 in 0-7
// and K {16..23}+(lane>>4)*8 in 8-15 (same order as the A fragment).
__device__ __forceinline__ v16bf load_b_frag(const bf16_t* __restrict__ wtile,
                                             int ld, int lane) {
  const int kofs = (lane >> 4) * 8;
  const int c    = lane & 15;
  const bf16_t* p = wtile + (size_t)c * ld + kofs;
  v8bf lo = *(const v8bf*)(p);
  v8bf hi = *(const v8bf*)(p + 16);
  v16bf out;
#pragma unroll
  for (int i = 0; i < 8; ++i) { out[i] = lo[i]; out[i + 8] = hi[i]; }
  return out;
}

__device__ __forceinline__ v8f wmma_bf16(v16bf a, v16bf b, v8f c) {
  return __builtin_amdgcn_wmma_f32_16x16x32_bf16(
      /*neg_a=*/false, a, /*neg_b=*/false, b,
      /*c_mod=*/(short)0, c, /*reuse_a=*/false, /*reuse_b=*/false);
}

// Async DMA of one 16-byte chunk: global memory -> LDS (ASYNCcnt-tracked).
// CDNA5 ISA: GLOBAL_LOAD_ASYNC_TO_LDS_B128, GV addressing.
// lds_off is the allocation-relative LDS byte address (= flat addr[31:0]).
__device__ __forceinline__ void async_g2l_b128(unsigned lds_off,
                                               const void* gptr) {
  asm volatile("global_load_async_to_lds_b128 %0, %1, off"
               :: "v"(lds_off), "v"(gptr) : "memory");
}
__device__ __forceinline__ void wait_async0() {
  asm volatile("s_wait_asynccnt 0" ::: "memory");
}
__device__ __forceinline__ unsigned lds_addr32(const void* p) {
  return (unsigned)(size_t)p;   // flat LDS addr low 32 bits = LDS offset
}

// ---------------------------------------------------------------------------
// Elementwise / data movement kernels
// ---------------------------------------------------------------------------

__global__ __launch_bounds__(256) void f32_to_bf16_kernel(
    const float* __restrict__ in, bf16_t* __restrict__ out, size_t n) {
  size_t i = (size_t)blockIdx.x * blockDim.x + threadIdx.x;
  size_t stride = (size_t)gridDim.x * blockDim.x;
  for (; i < n; i += stride) out[i] = f32_to_bf16_rn(in[i]);
}

__global__ __launch_bounds__(256) void init_h_kernel(
    const float* __restrict__ src, float* __restrict__ hf,
    bf16_t* __restrict__ hb, int n) {
  int i = blockIdx.x * blockDim.x + threadIdx.x;
  if (i < n) { float v = src[i]; hf[i] = v; hb[i] = f32_to_bf16_rn(v); }
}

// xs[t,b,:] = bf16(embed[tokens[t,b],:]) ; one block per (t,b)
__global__ __launch_bounds__(256) void embed_gather_kernel(
    const int* __restrict__ tok, const float* __restrict__ embed,
    bf16_t* __restrict__ xs) {
  const int row = blockIdx.x;                       // t*BATCH + b
  const int t   = tok[row];
  const float* src = embed + (size_t)t * EDIM;
  bf16_t*      dst = xs + (size_t)row * EDIM;
  for (int e = threadIdx.x; e < EDIM; e += blockDim.x)
    dst[e] = f32_to_bf16_rn(src[e]);
}

// ---------------------------------------------------------------------------
// Generic TN GEMM: C[M,N](f32) = A[M,K](bf16) @ W[N,K](bf16)^T + bias[N]
// Block = 256 threads (8 waves), block tile 64(M) x 128(N), wave tile 32x32.
// A (64x32) and B (128x32) k-tiles are staged in LDS via async global->LDS
// DMA, double buffered; WMMA operands come from LDS (rows padded to 40 elems
// = 20 banks to spread bank access). Requires M%64==0, N%128==0, K%32==0.
// ---------------------------------------------------------------------------
__global__ __launch_bounds__(256) void gemm_bf16_tn(
    const bf16_t* __restrict__ A, const bf16_t* __restrict__ W,
    const float* __restrict__ bias, float* __restrict__ C,
    int M, int N, int K) {
  __shared__ alignas(16) bf16_t sA[2][64][40];      //  5 KB x2
  __shared__ alignas(16) bf16_t sB[2][128][40];     // 10 KB x2
  (void)M;

  const int tid  = threadIdx.x;
  const int lane = tid & 31;
  const int wave = tid >> 5;
  const int mh   = wave & 1;                        // m-group: 0..1
  const int nh   = wave >> 1;                       // n-group: 0..3
  const int m0   = blockIdx.y * 64;
  const int n0   = blockIdx.x * 128;

  // DMA chunk mapping: 16 B per chunk. A: 64 rows x 4 chunks = 256 chunks
  // (one per thread). B: 128 rows x 4 chunks = 512 chunks (two per thread).
  const int crow = tid >> 2;                        // 0..63
  const int cofs = (tid & 3) * 8;                   // element offset 0/8/16/24

  v8f acc00 = {}, acc01 = {}, acc10 = {}, acc11 = {};

  auto issue = [&](int buf, int k) {
    async_g2l_b128(lds_addr32(&sA[buf][crow][cofs]),
                   A + (size_t)(m0 + crow) * K + k + cofs);
    async_g2l_b128(lds_addr32(&sB[buf][crow][cofs]),
                   W + (size_t)(n0 + crow) * K + k + cofs);
    async_g2l_b128(lds_addr32(&sB[buf][64 + crow][cofs]),
                   W + (size_t)(n0 + 64 + crow) * K + k + cofs);
  };

  issue(0, 0);
  const int NK = K / TILE_K;
  for (int kt = 0; kt < NK; ++kt) {
    const int cur = kt & 1;
    wait_async0();                                  // my DMA done
    __syncthreads();                                // everyone's DMA done
    if (kt + 1 < NK) issue(cur ^ 1, (kt + 1) * TILE_K);

    v16bf a0 = load_a_frag(&sA[cur][mh * 32][0],      40, lane);
    v16bf a1 = load_a_frag(&sA[cur][mh * 32 + 16][0], 40, lane);
    v16bf b0 = load_b_frag(&sB[cur][nh * 32][0],      40, lane);
    v16bf b1 = load_b_frag(&sB[cur][nh * 32 + 16][0], 40, lane);
    acc00 = wmma_bf16(a0, b0, acc00);
    acc01 = wmma_bf16(a0, b1, acc01);
    acc10 = wmma_bf16(a1, b0, acc10);
    acc11 = wmma_bf16(a1, b1, acc11);
    __syncthreads();                                // reads done before overwrite
  }

  const int wm0  = m0 + mh * 32;
  const int wn0  = n0 + nh * 32;
  const int half = lane >> 4;
  const int c    = lane & 15;
  const float bb0 = bias ? bias[wn0 + c]      : 0.0f;
  const float bb1 = bias ? bias[wn0 + 16 + c] : 0.0f;
#pragma unroll
  for (int i = 0; i < 8; ++i) {
    const int r = i + half * 8;
    C[(size_t)(wm0 + r)      * N + wn0 + c]      = acc00[i] + bb0;
    C[(size_t)(wm0 + r)      * N + wn0 + 16 + c] = acc01[i] + bb1;
    C[(size_t)(wm0 + 16 + r) * N + wn0 + c]      = acc10[i] + bb0;
    C[(size_t)(wm0 + 16 + r) * N + wn0 + 16 + c] = acc11[i] + bb1;
  }
}

// ---------------------------------------------------------------------------
// GRU recurrence, phase A:  z = sigmoid(xz + h@Uz^T), r = sigmoid(xr + h@Ur^T)
// Writes z (f32) and bf16(r*h) for phase B.  xz/xr already contain the bias
// (folded into the time-parallel input-projection GEMM).
// M = BATCH = 32 (2 m-tiles), block covers 128 N columns (wave -> 16 cols).
// Grid = HDIM/128 = 8 blocks.  Latency-bound: direct global loads (weights
// are L2-resident; no reuse within a step worth staging).
// ---------------------------------------------------------------------------
__global__ __launch_bounds__(256) void gru_step_zr(
    const bf16_t* __restrict__ h_bf, const float* __restrict__ h_f,
    const bf16_t* __restrict__ Uz,   const bf16_t* __restrict__ Ur,
    const float* __restrict__ xz,    const float* __restrict__ xr,
    float* __restrict__ zbuf,        bf16_t* __restrict__ rh_bf) {
  const int lane = threadIdx.x & 31;
  const int wave = threadIdx.x >> 5;
  const int n0   = blockIdx.x * 128 + wave * 16;
  const int K    = HDIM;

  v8f az0 = {}, az1 = {}, ar0 = {}, ar1 = {};
  const bf16_t* uzp = Uz + (size_t)n0 * K;
  const bf16_t* urp = Ur + (size_t)n0 * K;

  for (int k = 0; k < K; k += 32) {
    v16bf a0 = load_a_frag(h_bf + k, K, lane);                  // rows 0..15
    v16bf a1 = load_a_frag(h_bf + (size_t)16 * K + k, K, lane); // rows 16..31
    v16bf bz = load_b_frag(uzp + k, K, lane);
    v16bf br = load_b_frag(urp + k, K, lane);
    az0 = wmma_bf16(a0, bz, az0);
    az1 = wmma_bf16(a1, bz, az1);
    ar0 = wmma_bf16(a0, br, ar0);
    ar1 = wmma_bf16(a1, br, ar1);
  }

  const int half = lane >> 4;
  const int c    = lane & 15;
  const int col  = n0 + c;
#pragma unroll
  for (int i = 0; i < 8; ++i) {
    {
      const int row = i + half * 8;                 // batch rows 0..15
      const size_t idx = (size_t)row * HDIM + col;
      float z = sigmoid_f(xz[idx] + az0[i]);
      float r = sigmoid_f(xr[idx] + ar0[i]);
      zbuf[idx]  = z;
      rh_bf[idx] = f32_to_bf16_rn(r * h_f[idx]);
    }
    {
      const int row = 16 + i + half * 8;            // batch rows 16..31
      const size_t idx = (size_t)row * HDIM + col;
      float z = sigmoid_f(xz[idx] + az1[i]);
      float r = sigmoid_f(xr[idx] + ar1[i]);
      zbuf[idx]  = z;
      rh_bf[idx] = f32_to_bf16_rn(r * h_f[idx]);
    }
  }
}

// ---------------------------------------------------------------------------
// GRU recurrence, phase B:  hh = tanh(xh + (r*h)@Uh^T);
//                           h' = z*hh + (1-z)*h
// Writes h' as f32 (state), bf16 (next-step operand), bf16 into the layer's
// output sequence, and (last step only) into the `finals` output.
// ---------------------------------------------------------------------------
__global__ __launch_bounds__(256) void gru_step_h(
    const bf16_t* __restrict__ rh_bf, const float* __restrict__ h_f,
    const bf16_t* __restrict__ Uh,    const float* __restrict__ xh,
    const float* __restrict__ zbuf,
    float* __restrict__ h_f_out,      bf16_t* __restrict__ h_bf_out,
    bf16_t* __restrict__ hs_bf_out,   float* __restrict__ final_out,
    int write_final) {
  const int lane = threadIdx.x & 31;
  const int wave = threadIdx.x >> 5;
  const int n0   = blockIdx.x * 128 + wave * 16;
  const int K    = HDIM;

  v8f ah0 = {}, ah1 = {};
  const bf16_t* uhp = Uh + (size_t)n0 * K;

  for (int k = 0; k < K; k += 32) {
    v16bf a0 = load_a_frag(rh_bf + k, K, lane);
    v16bf a1 = load_a_frag(rh_bf + (size_t)16 * K + k, K, lane);
    v16bf bh = load_b_frag(uhp + k, K, lane);
    ah0 = wmma_bf16(a0, bh, ah0);
    ah1 = wmma_bf16(a1, bh, ah1);
  }

  const int half = lane >> 4;
  const int c    = lane & 15;
  const int col  = n0 + c;
#pragma unroll
  for (int i = 0; i < 8; ++i) {
#pragma unroll
    for (int mt = 0; mt < 2; ++mt) {
      const int row = mt * 16 + i + half * 8;
      const size_t idx = (size_t)row * HDIM + col;
      const float acc = (mt == 0) ? ah0[i] : ah1[i];
      float hh = tanh_f(xh[idx] + acc);
      float z  = zbuf[idx];
      float hp = h_f[idx];
      float hn = z * hh + (1.0f - z) * hp;
      h_f_out[idx] = hn;
      bf16_t hb = f32_to_bf16_rn(hn);
      h_bf_out[idx]  = hb;
      hs_bf_out[idx] = hb;
      if (write_final) final_out[idx] = hn;
    }
  }
}

// ---------------------------------------------------------------------------
// Host orchestration
// ---------------------------------------------------------------------------
extern "C" void kernel_launch(void* const* d_in, const int* in_sizes, int n_in,
                              void* d_out, int out_size, void* d_ws, size_t ws_size,
                              hipStream_t stream) {
  (void)in_sizes; (void)n_in; (void)out_size; (void)ws_size;

  const int*   tok    = (const int*)  d_in[0];   // (T,B)
  const float* hidden = (const float*)d_in[1];   // (L,B,H)
  const float* embed  = (const float*)d_in[2];   // (V,E)
  const float* Wf[2][3] = {{(const float*)d_in[3],  (const float*)d_in[6],  (const float*)d_in[9]},
                           {(const float*)d_in[12], (const float*)d_in[15], (const float*)d_in[18]}};
  const float* Uf[2][3] = {{(const float*)d_in[4],  (const float*)d_in[7],  (const float*)d_in[10]},
                           {(const float*)d_in[13], (const float*)d_in[16], (const float*)d_in[19]}};
  const float* bf[2][3] = {{(const float*)d_in[5],  (const float*)d_in[8],  (const float*)d_in[11]},
                           {(const float*)d_in[14], (const float*)d_in[17], (const float*)d_in[20]}};
  const float* linW = (const float*)d_in[21];    // (V,H)
  const float* linb = (const float*)d_in[22];    // (V,)

  float* logits = (float*)d_out;                               // (T,B,V)
  float* finals = logits + (size_t)T_STEPS * BATCH * VOCAB;    // (L,B,H)

  // -------- workspace carve-out (all 256B aligned) --------
  char* wsp = (char*)d_ws;
  auto alloc = [&](size_t bytes) -> char* {
    char* p = wsp;
    wsp += (bytes + 255) & ~(size_t)255;
    return p;
  };
  const size_t HH  = (size_t)HDIM * HDIM;
  const size_t TBH = (size_t)T_STEPS * BATCH * HDIM;

  bf16_t* Wb[2][3]; bf16_t* Ub[2][3];
  for (int l = 0; l < 2; ++l)
    for (int g = 0; g < 3; ++g) {
      Wb[l][g] = (bf16_t*)alloc(HH * sizeof(bf16_t));
      Ub[l][g] = (bf16_t*)alloc(HH * sizeof(bf16_t));
    }
  bf16_t* linWb = (bf16_t*)alloc((size_t)VOCAB * HDIM * sizeof(bf16_t));
  bf16_t* xs_bf = (bf16_t*)alloc((size_t)T_STEPS * BATCH * EDIM * sizeof(bf16_t));
  float*  xw[3];
  for (int g = 0; g < 3; ++g) xw[g] = (float*)alloc(TBH * sizeof(float));
  bf16_t* hs[2];
  hs[0] = (bf16_t*)alloc(TBH * sizeof(bf16_t));
  hs[1] = (bf16_t*)alloc(TBH * sizeof(bf16_t));
  float*  h_f  = (float*) alloc((size_t)BATCH * HDIM * sizeof(float));
  bf16_t* h_b  = (bf16_t*)alloc((size_t)BATCH * HDIM * sizeof(bf16_t));
  float*  zb   = (float*) alloc((size_t)BATCH * HDIM * sizeof(float));
  bf16_t* rh   = (bf16_t*)alloc((size_t)BATCH * HDIM * sizeof(bf16_t));

  // -------- 1. weight conversion to bf16 --------
  auto cv = [&](const float* src, bf16_t* dst, size_t n) {
    f32_to_bf16_kernel<<<2048, 256, 0, stream>>>(src, dst, n);
  };
  for (int l = 0; l < 2; ++l)
    for (int g = 0; g < 3; ++g) { cv(Wf[l][g], Wb[l][g], HH); cv(Uf[l][g], Ub[l][g], HH); }
  cv(linW, linWb, (size_t)VOCAB * HDIM);

  // -------- 2. embedding gather (f32 -> bf16) --------
  embed_gather_kernel<<<T_STEPS * BATCH, 256, 0, stream>>>(tok, embed, xs_bf);

  auto gemm = [&](const bf16_t* A, const bf16_t* W, const float* bias,
                  float* C, int M, int N, int K) {
    dim3 grid(N / 128, M / 64);
    gemm_bf16_tn<<<grid, 256, 0, stream>>>(A, W, bias, C, M, N, K);
  };

  const int MB = T_STEPS * BATCH;   // 4096

  // -------- 3. two GRU layers --------
  const bf16_t* layer_in = xs_bf;
  int in_K = EDIM;
  for (int l = 0; l < 2; ++l) {
    // time-parallel input projections (bias folded in here)
    for (int g = 0; g < 3; ++g)
      gemm(layer_in, Wb[l][g], bf[l][g], xw[g], MB, HDIM, in_K);

    // init recurrent state from hidden[l]
    init_h_kernel<<<(BATCH * HDIM + 255) / 256, 256, 0, stream>>>(
        hidden + (size_t)l * BATCH * HDIM, h_f, h_b, BATCH * HDIM);

    // sequential recurrence: 2 kernels per step ((r*h)@Uh dependency)
    for (int t = 0; t < T_STEPS; ++t) {
      const size_t off = (size_t)t * BATCH * HDIM;
      gru_step_zr<<<HDIM / 128, 256, 0, stream>>>(
          h_b, h_f, Ub[l][0], Ub[l][1], xw[0] + off, xw[1] + off, zb, rh);
      gru_step_h<<<HDIM / 128, 256, 0, stream>>>(
          rh, h_f, Ub[l][2], xw[2] + off, zb,
          h_f, h_b, hs[l] + off,
          finals + (size_t)l * BATCH * HDIM, (t == T_STEPS - 1) ? 1 : 0);
    }
    layer_in = hs[l];
    in_K = HDIM;
  }

  // -------- 4. logits = hs1 @ linW^T + linb  (the 268-GFLOP GEMM) --------
  gemm(hs[1], linWb, linb, logits, MB, VOCAB, HDIM);
}
